// DecoderAutoregSALN_90658169684689
// MI455X (gfx1250) — matrix operand order repaired and checked by reading, not compiled, not run on hardware
//
// MI455X (gfx1250) fused autoregressive FaceFormer decoder.
//
// Strategy:
//  * cross-attn mask is an exact identity -> precompute (content@Wv+bv)@Wo+bo once
//  * causal structure -> single incremental pass with self-attn KV cache
//  * bf16 WMMA (v_wmma_f32_16x16x32_bf16) for every GEMM, fp32 accumulation
//  * sequential tail = persistent 1024-thread block (32 wave32 waves, s_barrier sync)
//  * parallel phase = weight bf16+transpose, cross-attn precompute, SALN gamma/beta
//  * invalid A-rows read a zeroed pad buffer (pointer select, no per-iter EXEC masking)
// Workspace use: ~30 MB (weights in bf16 + caches), well inside L2 (192 MB).

#include <hip/hip_runtime.h>

typedef __bf16 bf16_t;
typedef bf16_t v16bf __attribute__((ext_vector_type(16)));
typedef float  v8f   __attribute__((ext_vector_type(8)));
typedef unsigned int v4u __attribute__((ext_vector_type(4)));
typedef unsigned int v8u __attribute__((ext_vector_type(8)));

#define DD   512
#define FFD  2048
#define NF   32
#define NB   2
#define NH   8
#define HDM  64
#define NL   4
#define MM   256
#define ZPAD_ELEMS 2080   // >= max K (2048) + fragment overreach

__device__ __forceinline__ unsigned short f2bf(float x) {
  unsigned u = __builtin_bit_cast(unsigned, x);
  unsigned r = u + 0x7fffu + ((u >> 16) & 1u);   // round-to-nearest-even
  return (unsigned short)(r >> 16);
}

// A-matrix 16x32 bf16 fragment (ISA 7.12.2): lane = row M; lanes 0-15 hold
// K=[0..8)+[16..24), lanes 16-31 hold K=[8..16)+[24..32) -> two 16B chunks.
__device__ __forceinline__ v16bf load_a_frag(const unsigned short* row, int hi) {
  v4u lo = *(const v4u*)(row + hi * 8);
  v4u hh = *(const v4u*)(row + 16 + hi * 8);
  v8u r;
  r[0] = lo[0]; r[1] = lo[1]; r[2] = lo[2]; r[3] = lo[3];
  r[4] = hh[0]; r[5] = hh[1]; r[6] = hh[2]; r[7] = hh[3];
  return __builtin_bit_cast(v16bf, r);
}

// B-matrix 32x16 bf16 fragment: lane&15 = column N; K order matches A fragment:
// lanes 0-15 hold K=[0..8)+[16..24), lanes 16-31 hold K=[8..16)+[24..32).
__device__ __forceinline__ v16bf load_b_frag(const unsigned short* col, int hi) {
  v4u lo = *(const v4u*)(col + hi * 8);
  v4u hh = *(const v4u*)(col + 16 + hi * 8);
  v8u r;
  r[0] = lo[0]; r[1] = lo[1]; r[2] = lo[2]; r[3] = lo[3];
  r[4] = hh[0]; r[5] = hh[1]; r[6] = hh[2]; r[7] = hh[3];
  return __builtin_bit_cast(v16bf, r);
}

__device__ __forceinline__ v8f wmma_bf16(v16bf a, v16bf b, v8f c) {
  return __builtin_amdgcn_wmma_f32_16x16x32_bf16(false, a, false, b, (short)0, c,
                                                 false, false);
}

// One 16x16 output tile: A (<=16 rows, bf16, row-major, lda elems) x Wt (N x K bf16).
// Rows >= validRows read from the zeroed pad: unconditional loads, no EXEC churn.
__device__ __forceinline__ v8f tile_gemm(const unsigned short* A, int lda, int validRows,
                                         const unsigned short* __restrict__ W, int K,
                                         int ntile, int lane,
                                         const unsigned short* zpad) {
  const int r = lane & 15, hi = lane >> 4;
  const unsigned short* a_row = (r < validRows) ? (A + (long)r * lda) : zpad;
  const unsigned short* w_col = W + (long)(ntile * 16 + r) * K;
  v8f acc = {0.f,0.f,0.f,0.f,0.f,0.f,0.f,0.f};
  for (int kk = 0; kk < K; kk += 32) {
    __builtin_prefetch(w_col + kk + 256, 0, 3);   // near-cache prefetch of next chunk
    v16bf a = load_a_frag(a_row + kk, hi);
    v16bf b = load_b_frag(w_col + kk, hi);
    acc = wmma_bf16(a, b, acc);
  }
  return acc;
}

// ---------------------------------------------------------------------------
// Phase-A kernels (parallel preprocessing)
// ---------------------------------------------------------------------------

// fp32 (K x N, batched) -> bf16 transposed (N x K)
__global__ void conv_t_kernel(const float* __restrict__ W, unsigned short* __restrict__ Wt,
                              int K, int N) {
  long per = (long)K * N;
  const float* w = W + per * blockIdx.z;
  unsigned short* o = Wt + per * blockIdx.z;
  long idx = (long)blockIdx.x * blockDim.x + threadIdx.x;
  if (idx < per) {
    int k = (int)(idx / N), n = (int)(idx % N);
    o[(long)n * K + k] = f2bf(w[idx]);
  }
}

__global__ void conv_b16_kernel(const float* __restrict__ x, unsigned short* __restrict__ y,
                                int n) {
  int i = blockIdx.x * blockDim.x + threadIdx.x;
  if (i < n) y[i] = f2bf(x[i]);
}

// SALN gamma/beta: GB[l][j][b][0..2D) = style[b] @ saln_W[l][j] + saln_b[l][j]
__global__ void gb_kernel(const float* __restrict__ style, const float* __restrict__ W,
                          const float* __restrict__ bsa, float* __restrict__ GB) {
  int idx = blockIdx.x * blockDim.x + threadIdx.x;   // L*3*B*2D = 24576
  if (idx >= NL * 3 * NB * 2 * DD) return;
  int n = idx & 1023;
  int b = (idx >> 10) & 1;
  int j = (idx >> 11) % 3;
  int l = idx / (1024 * 2 * 3);
  float acc = bsa[(l * 3 + j) * 1024 + n];
  const float* wp = W + (((long)l * 3 + j) * DD) * 1024 + n;
  const float* sp = style + b * DD;
  for (int k = 0; k < DD; ++k) acc += sp[k] * wp[(long)k * 1024];
  GB[idx] = acc;   // layout ((l*3+j)*2 + b)*1024 + n == idx
}

// Generic batched WMMA GEMM: C = A(bf16, Mrows x K) @ Wt(bf16, N x K) + bias
__global__ void wmma_gemm_kernel(const unsigned short* __restrict__ A, int lda, long aB,
                                 const unsigned short* __restrict__ W, long wB,
                                 const float* __restrict__ bias, long biasB,
                                 float* Cf, unsigned short* Cbf, int ldc, long cB,
                                 int Mrows, int K, int relu,
                                 const unsigned short* __restrict__ zpad) {
  const int lane = threadIdx.x & 31;
  const int ntile = blockIdx.x, mtile = blockIdx.y, z = blockIdx.z;
  const int r = lane & 15, hi = lane >> 4;
  const unsigned short* a0 = A + (long)z * aB + (long)(mtile * 16) * lda;
  v8f acc = tile_gemm(a0, lda, Mrows - mtile * 16, W + (long)z * wB, K, ntile, lane, zpad);
  const int n = ntile * 16 + r;
  const float bz = bias ? bias[(long)z * biasB + n] : 0.f;
  const int rbase = mtile * 16 + hi * 8;
#pragma unroll
  for (int j = 0; j < 8; ++j) {
    int row = rbase + j;
    if (row < Mrows) {
      float v = acc[j] + bz;
      if (relu) v = fmaxf(v, 0.f);
      if (Cf)  Cf[(long)z * cB + (long)row * ldc + n] = v;
      if (Cbf) Cbf[(long)z * cB + (long)row * ldc + n] = f2bf(v);
    }
  }
}

// ---------------------------------------------------------------------------
// Phase-B: persistent sequential decoder (single WGP, 32 waves)
// ---------------------------------------------------------------------------

__device__ __forceinline__ float wave_sum(float v) {
  for (int m = 16; m; m >>= 1) v += __shfl_xor(v, m, 32);
  return v;
}

__device__ __forceinline__ float ppe_val(int t, int d) {
  int pos = t % 30;
  float div = __expf((float)(d & ~1) * (-9.210340371976184f / 512.f));
  float a = (float)pos * div;
  return (d & 1) ? __cosf(a) : __sinf(a);
}

__device__ __forceinline__ void saln_ln(const float* ys, float* xs, unsigned short* xb,
                                        const float* __restrict__ gb, float* red, int tid) {
  if (tid < 4) red[tid] = 0.f;
  __syncthreads();
  int b = tid >> 9, d = tid & 511;
  float v = ys[tid];
  float s = wave_sum(v);
  if ((tid & 31) == 0) atomicAdd(&red[b], s);   // ds_add_f32
  __syncthreads();
  float mu = red[b] * (1.f / 512.f);
  float dv = v - mu;
  float q = wave_sum(dv * dv);
  if ((tid & 31) == 0) atomicAdd(&red[2 + b], q);
  __syncthreads();
  float rs = rsqrtf(red[2 + b] * (1.f / 512.f) + 1e-5f);
  float o = dv * rs * gb[b * 1024 + d] + gb[b * 1024 + 512 + d];
  xs[tid] = o;
  xb[tid] = f2bf(o);
  __syncthreads();
}

__global__ __launch_bounds__(1024) void decode_persistent(
    const unsigned short* __restrict__ sqw, const unsigned short* __restrict__ skw,
    const unsigned short* __restrict__ svw, const unsigned short* __restrict__ sow,
    const unsigned short* __restrict__ f1w, const unsigned short* __restrict__ f2w,
    const unsigned short* __restrict__ mow, const unsigned short* __restrict__ mrw,
    const float* __restrict__ bq, const float* __restrict__ bk,
    const float* __restrict__ bv, const float* __restrict__ bo,
    const float* __restrict__ b1, const float* __restrict__ b2,
    const float* __restrict__ mob, const float* __restrict__ mrb,
    const float* __restrict__ GB, const float* __restrict__ CA,
    const float* __restrict__ emb0, float* kc, float* vc, float* out) {
  __shared__ __align__(32) float xs[2 * DD], ys[2 * DD], qs[2 * DD];
  __shared__ __align__(32) unsigned short xb[2 * DD], ob[2 * DD];
  __shared__ __align__(32) unsigned short h1b[2 * FFD], orb[2 * MM];
  __shared__ __align__(32) unsigned short zp[ZPAD_ELEMS];
  __shared__ float red[4];

  const int tid = threadIdx.x, w = tid >> 5, lane = tid & 31;
  for (int i = tid; i < ZPAD_ELEMS; i += 1024) zp[i] = 0;   // shared zero pad
  {
    int b = tid >> 9, d = tid & 511;
    float v = emb0[b * DD + d] + ppe_val(0, d);
    xs[tid] = v;
    xb[tid] = f2bf(v);
  }
  __syncthreads();

  for (int t = 0; t < NF; ++t) {
    for (int l = 0; l < NL; ++l) {
      // -------- fused Q/K/V projection: 96 tiles over 32 waves --------
      for (int tt = w; tt < 96; tt += 32) {
        int which = tt >> 5, nt = tt & 31;
        const unsigned short* Wt = (which == 0) ? sqw : (which == 1) ? skw : svw;
        const float* bb = (which == 0) ? bq : (which == 1) ? bk : bv;
        v8f acc = tile_gemm(xb, DD, 2, Wt + (long)l * DD * DD, DD, nt, lane, zp);
        if (lane < 16) {
          int n = nt * 16 + lane;
          float r0 = acc[0] + bb[l * DD + n];
          float r1 = acc[1] + bb[l * DD + n];
          if (which == 0) { qs[n] = r0; qs[DD + n] = r1; }
          else {
            float* c = (which == 1) ? kc : vc;
            c[(((long)l * NB + 0) * NF + t) * DD + n] = r0;
            c[(((long)l * NB + 1) * NF + t) * DD + n] = r1;
          }
        }
      }
      __threadfence_block();
      __syncthreads();

      // -------- causal self-attention with KV cache (waves 0..15) --------
      if (w < 16) {
        int b = w >> 3, h = w & 7;
        const float* kcb = kc + (((long)l * NB + b) * NF) * DD + h * HDM;
        const float* vcb = vc + (((long)l * NB + b) * NF) * DD + h * HDM;
        float sc = -1e30f;
        if (lane <= t) {
          const float* kv = kcb + (long)lane * DD;
          float acc = 0.f;
#pragma unroll 8
          for (int e = 0; e < HDM; ++e) acc += qs[b * DD + h * HDM + e] * kv[e];
          float slope = exp2f(-(float)(h + 1));             // ALiBi slope 0.5^(h+1)
          sc = acc * 0.125f + slope * (-(float)((t - lane) / 30));
        }
        float m = sc;
        for (int msk = 16; msk; msk >>= 1) m = fmaxf(m, __shfl_xor(m, msk, 32));
        float p = (lane <= t) ? __expf(sc - m) : 0.f;
        float ssum = p;
        for (int msk = 16; msk; msk >>= 1) ssum += __shfl_xor(ssum, msk, 32);
        float att = p / ssum;
        float a0 = 0.f, a1 = 0.f;
        for (int s2 = 0; s2 <= t; ++s2) {
          float aw = __shfl(att, s2, 32);
          const float* vv = vcb + (long)s2 * DD;
          a0 += aw * vv[lane];
          a1 += aw * vv[32 + lane];
        }
        ob[b * DD + h * HDM + lane] = f2bf(a0);
        ob[b * DD + h * HDM + 32 + lane] = f2bf(a1);
      }
      __syncthreads();

      // -------- output projection + residual --------
      {
        v8f acc = tile_gemm(ob, DD, 2, sow + (long)l * DD * DD, DD, w, lane, zp);
        if (lane < 16) {
          int n = w * 16 + lane;
          ys[n]      = xs[n]      + acc[0] + bo[l * DD + n];
          ys[DD + n] = xs[DD + n] + acc[1] + bo[l * DD + n];
        }
      }
      __syncthreads();
      saln_ln(ys, xs, xb, GB + ((long)(l * 3 + 0) * 2) * 1024, red, tid);

      // -------- cross-attention (precomputed, mask == identity) --------
      {
        int b = tid >> 9, d = tid & 511;
        ys[tid] = xs[tid] + CA[(((long)l * NB + b) * NF + t) * DD + d];
      }
      __syncthreads();
      saln_ln(ys, xs, xb, GB + ((long)(l * 3 + 1) * 2) * 1024, red, tid);

      // -------- FFN --------
      for (int tt = w; tt < 128; tt += 32) {
        v8f acc = tile_gemm(xb, DD, 2, f1w + (long)l * FFD * DD, DD, tt, lane, zp);
        if (lane < 16) {
          int n = tt * 16 + lane;
          h1b[n]       = f2bf(fmaxf(acc[0] + b1[l * FFD + n], 0.f));
          h1b[FFD + n] = f2bf(fmaxf(acc[1] + b1[l * FFD + n], 0.f));
        }
      }
      __syncthreads();
      {
        v8f acc = tile_gemm(h1b, FFD, 2, f2w + (long)l * DD * FFD, FFD, w, lane, zp);
        if (lane < 16) {
          int n = w * 16 + lane;
          ys[n]      = xs[n]      + acc[0] + b2[l * DD + n];
          ys[DD + n] = xs[DD + n] + acc[1] + b2[l * DD + n];
        }
      }
      __syncthreads();
      saln_ln(ys, xs, xb, GB + ((long)(l * 3 + 2) * 2) * 1024, red, tid);
    }

    // -------- motion readout (final output row t) --------
    if (w < 16) {
      v8f acc = tile_gemm(xb, DD, 2, mrw, DD, w, lane, zp);
      if (lane < 16) {
        int n = w * 16 + lane;
        float r0 = acc[0] + mrb[n], r1 = acc[1] + mrb[n];
        out[(0 * NF + t) * MM + n] = r0;
        out[(1 * NF + t) * MM + n] = r1;
        orb[n] = f2bf(r0);
        orb[MM + n] = f2bf(r1);
      }
    }
    __syncthreads();

    // -------- next decoder embedding + PPE --------
    if (t + 1 < NF) {
      v8f acc = tile_gemm(orb, MM, 2, mow, MM, w, lane, zp);
      if (lane < 16) {
        int n = w * 16 + lane;
        float e0 = acc[0] + mob[n] + ppe_val(t + 1, n);
        float e1 = acc[1] + mob[n] + ppe_val(t + 1, n);
        xs[n] = e0;      xs[DD + n] = e1;
        xb[n] = f2bf(e0); xb[DD + n] = f2bf(e1);
      }
      __syncthreads();
    }
  }
}

// ---------------------------------------------------------------------------
// Host launch
// ---------------------------------------------------------------------------

extern "C" void kernel_launch(void* const* d_in, const int* in_sizes, int n_in,
                              void* d_out, int out_size, void* d_ws, size_t ws_size,
                              hipStream_t stream) {
  (void)in_sizes; (void)n_in; (void)out_size; (void)ws_size;
  char* ws = (char*)d_ws;
  size_t off = 0;
  auto take = [&](size_t bytes) {
    size_t r = off;
    off += (bytes + 255) & ~(size_t)255;
    return r;
  };

  const size_t szDD = (size_t)NL * DD * DD * 2;      // L x D x D bf16
  size_t o_sq = take(szDD), o_sk = take(szDD), o_sv = take(szDD), o_so = take(szDD);
  size_t o_f1 = take((size_t)NL * FFD * DD * 2);
  size_t o_f2 = take((size_t)NL * DD * FFD * 2);
  size_t o_cv = take(szDD), o_co = take(szDD);
  size_t o_mo = take((size_t)DD * MM * 2);           // motion_W^T  (512 x 256)
  size_t o_mr = take((size_t)MM * DD * 2);           // motionr_W^T (256 x 512)
  size_t o_cb = take((size_t)NB * NF * DD * 2);      // content bf16
  size_t o_ib = take((size_t)NB * MM * 2);           // init_state bf16
  size_t o_tb = take((size_t)NL * NB * NF * DD * 2); // cross-attn V bf16
  size_t o_CA = take((size_t)NL * NB * NF * DD * 4); // cross-attn contribution f32
  size_t o_GB = take((size_t)NL * 3 * NB * 2 * DD * 4);
  size_t o_e0 = take((size_t)NB * DD * 4);
  size_t o_kc = take((size_t)NL * NB * NF * DD * 4);
  size_t o_vc = take((size_t)NL * NB * NF * DD * 4);
  size_t o_zp = take((size_t)ZPAD_ELEMS * 2 + 64);   // global zero pad for generic GEMM

  const float* content  = (const float*)d_in[0];
  const float* style    = (const float*)d_in[1];
  const float* init_st  = (const float*)d_in[2];
  const float* motion_W = (const float*)d_in[3];
  const float* motion_b = (const float*)d_in[4];
  const float* motionr_W= (const float*)d_in[5];
  const float* motionr_b= (const float*)d_in[6];

  hipMemsetAsync(ws + o_zp, 0, (size_t)ZPAD_ELEMS * 2 + 64, stream);

  // weights -> bf16, transposed (N x K), batched over L
  conv_t_kernel<<<dim3(1024, 1, NL), 256, 0, stream>>>((const float*)d_in[7],  (unsigned short*)(ws + o_sq), DD, DD);
  conv_t_kernel<<<dim3(1024, 1, NL), 256, 0, stream>>>((const float*)d_in[9],  (unsigned short*)(ws + o_sk), DD, DD);
  conv_t_kernel<<<dim3(1024, 1, NL), 256, 0, stream>>>((const float*)d_in[11], (unsigned short*)(ws + o_sv), DD, DD);
  conv_t_kernel<<<dim3(1024, 1, NL), 256, 0, stream>>>((const float*)d_in[13], (unsigned short*)(ws + o_so), DD, DD);
  conv_t_kernel<<<dim3(1024, 1, NL), 256, 0, stream>>>((const float*)d_in[19], (unsigned short*)(ws + o_cv), DD, DD);
  conv_t_kernel<<<dim3(1024, 1, NL), 256, 0, stream>>>((const float*)d_in[21], (unsigned short*)(ws + o_co), DD, DD);
  conv_t_kernel<<<dim3(4096, 1, NL), 256, 0, stream>>>((const float*)d_in[23], (unsigned short*)(ws + o_f1), DD, FFD);
  conv_t_kernel<<<dim3(4096, 1, NL), 256, 0, stream>>>((const float*)d_in[25], (unsigned short*)(ws + o_f2), FFD, DD);
  conv_t_kernel<<<dim3(512, 1, 1),   256, 0, stream>>>(motion_W,  (unsigned short*)(ws + o_mo), MM, DD);
  conv_t_kernel<<<dim3(512, 1, 1),   256, 0, stream>>>(motionr_W, (unsigned short*)(ws + o_mr), DD, MM);

  conv_b16_kernel<<<128, 256, 0, stream>>>(content, (unsigned short*)(ws + o_cb), NB * NF * DD);
  conv_b16_kernel<<<2, 256, 0, stream>>>(init_st, (unsigned short*)(ws + o_ib), NB * MM);

  gb_kernel<<<96, 256, 0, stream>>>(style, (const float*)d_in[27], (const float*)d_in[28],
                                    (float*)(ws + o_GB));

  const unsigned short* zpad = (const unsigned short*)(ws + o_zp);

  // emb0 = init_state @ motion_W + motion_b  (rows = 2)
  wmma_gemm_kernel<<<dim3(DD / 16, 1, 1), 32, 0, stream>>>(
      (const unsigned short*)(ws + o_ib), MM, 0,
      (const unsigned short*)(ws + o_mo), 0,
      motion_b, 0,
      (float*)(ws + o_e0), nullptr, DD, 0, NB, MM, 0, zpad);

  // cross-attn (identity mask): tmp = content @ Wv + bv ; CA = tmp @ Wo + bo
  wmma_gemm_kernel<<<dim3(DD / 16, 4, NL), 32, 0, stream>>>(
      (const unsigned short*)(ws + o_cb), DD, 0,
      (const unsigned short*)(ws + o_cv), (long)DD * DD,
      (const float*)d_in[20], DD,
      nullptr, (unsigned short*)(ws + o_tb), DD, (long)NB * NF * DD, NB * NF, DD, 0, zpad);
  wmma_gemm_kernel<<<dim3(DD / 16, 4, NL), 32, 0, stream>>>(
      (const unsigned short*)(ws + o_tb), DD, (long)NB * NF * DD,
      (const unsigned short*)(ws + o_co), (long)DD * DD,
      (const float*)d_in[22], DD,
      (float*)(ws + o_CA), nullptr, DD, (long)NB * NF * DD, NB * NF, DD, 0, zpad);

  // sequential incremental decode: 1 persistent workgroup, 32 waves
  decode_persistent<<<1, 1024, 0, stream>>>(
      (const unsigned short*)(ws + o_sq), (const unsigned short*)(ws + o_sk),
      (const unsigned short*)(ws + o_sv), (const unsigned short*)(ws + o_so),
      (const unsigned short*)(ws + o_f1), (const unsigned short*)(ws + o_f2),
      (const unsigned short*)(ws + o_mo), (const unsigned short*)(ws + o_mr),
      (const float*)d_in[8], (const float*)d_in[10], (const float*)d_in[12],
      (const float*)d_in[14], (const float*)d_in[24], (const float*)d_in[26],
      motion_b, motionr_b,
      (const float*)(ws + o_GB), (const float*)(ws + o_CA), (const float*)(ws + o_e0),
      (float*)(ws + o_kc), (float*)(ws + o_vc), (float*)d_out);
}
